// Backbone_56607668961933
// MI455X (gfx1250) — matrix operand order repaired and checked by reading, not compiled, not run on hardware
//
#include <hip/hip_runtime.h>
#include <hip/hip_bf16.h>
#include <math.h>

#define HID 128

static const int N_AGENT = 64, T_HIST = 50, N_MODES = 6, T_FUT = 60;
static const int NT  = N_AGENT * T_HIST;        // 3200
static const int NTK = NT * N_MODES;            // 19200
static const int N_LANE = 768;
static const int E_T2M = 128000, E_L2M = 150000, E_H = 120000, E_A = 120000, E_S = 115200;
static const int EMAX = 150000;

typedef __attribute__((ext_vector_type(16))) __bf16 v16bf;
typedef __attribute__((ext_vector_type(8)))  __bf16 v8bf;
typedef __attribute__((ext_vector_type(8)))  float  v8f;
typedef __attribute__((ext_vector_type(4)))  float  v4f;

// ---------------------------------------------------------------- utilities

__device__ inline float warp_sum(float v) {
#pragma unroll
  for (int m = 16; m >= 1; m >>= 1) v += __shfl_xor(v, m, 32);
  return v;
}

__device__ inline void atomicMaxFloat(float* addr, float val) {
  if (val >= 0.0f) atomicMax((int*)addr, __float_as_int(val));
  else             atomicMin((unsigned int*)addr, __float_as_uint(val));
}

// ---------------------------------------------------------------- fill

__global__ void fill_kernel(float* __restrict__ p, float v, long n) {
  long i = (long)blockIdx.x * blockDim.x + threadIdx.x;
  if (i < n) p[i] = v;
}

// ---------------------------------------------------------------- weight pack: W[K,N] f32 -> Wp bf16 in WMMA-B fragment order
// Wp[(k/32)*32*N + n*32 + (k%32)]; a B fragment (given kb,col,hi) is 16 contiguous bf16.

__global__ void pack_w_kernel(const float* __restrict__ W, __bf16* __restrict__ Wp,
                              int K, int N) {
  long i = (long)blockIdx.x * blockDim.x + threadIdx.x;
  if (i >= (long)K * N) return;
  int k = (int)(i / N), n = (int)(i % N);
  Wp[(long)(k >> 5) * 32 * N + (long)n * 32 + (k & 31)] = (__bf16)W[i];
}

// ---------------------------------------------------------------- mode token broadcast

__global__ void mode_bcast_kernel(const float* __restrict__ tok, float* __restrict__ m, long total) {
  long i = (long)blockIdx.x * blockDim.x + threadIdx.x;
  if (i < total) {
    long node = i >> 7;
    int  c    = (int)(i & 127);
    m[i] = tok[(node % N_MODES) * HID + c];
  }
}

// ---------------------------------------------------------------- fused Linear(fi small) + LN + ReLU (wave per row)

__global__ void mlp_l1_small_kernel(const float* __restrict__ X, int fi,
                                    const float* __restrict__ W1, const float* __restrict__ b1,
                                    const float* __restrict__ g,  const float* __restrict__ bn,
                                    float* __restrict__ H, int rows) {
  int wave = threadIdx.x >> 5, lane = threadIdx.x & 31;
  int row = blockIdx.x * 8 + wave;
  if (row >= rows) return;
  const float* x = X + (long)row * fi;
  float xc[8];
  for (int i = 0; i < fi && i < 8; i++) xc[i] = x[i];
  float acc[4];
#pragma unroll
  for (int j = 0; j < 4; j++) {
    int c = lane + 32 * j;
    float a = b1[c];
    for (int i = 0; i < fi && i < 8; i++) a += xc[i] * W1[i * HID + c];
    acc[j] = a;
  }
  float s = 0.f, ss = 0.f;
#pragma unroll
  for (int j = 0; j < 4; j++) { s += acc[j]; ss += acc[j] * acc[j]; }
  s = warp_sum(s); ss = warp_sum(ss);
  float mean = s * (1.0f / 128.0f);
  float var  = ss * (1.0f / 128.0f) - mean * mean;
  float inv  = rsqrtf(var + 1e-5f);
#pragma unroll
  for (int j = 0; j < 4; j++) {
    int c = lane + 32 * j;
    float h = (acc[j] - mean) * inv * g[c] + bn[c];
    H[(long)row * HID + c] = h > 0.f ? h : 0.f;
  }
}

// ---------------------------------------------------------------- LayerNorm (wave per row), optional ReLU

__global__ void ln_kernel(const float* __restrict__ X, const float* __restrict__ g,
                          const float* __restrict__ b, float* __restrict__ Y,
                          int rows, int relu) {
  int wave = threadIdx.x >> 5, lane = threadIdx.x & 31;
  int row = blockIdx.x * 8 + wave;
  if (row >= rows) return;
  const float* x = X + (long)row * HID;
  float v[4];
  float s = 0.f, ss = 0.f;
#pragma unroll
  for (int j = 0; j < 4; j++) {
    v[j] = x[lane + 32 * j];
    s += v[j]; ss += v[j] * v[j];
  }
  s = warp_sum(s); ss = warp_sum(ss);
  float mean = s * (1.0f / 128.0f);
  float inv  = rsqrtf(ss * (1.0f / 128.0f) - mean * mean + 1e-5f);
#pragma unroll
  for (int j = 0; j < 4; j++) {
    int c = lane + 32 * j;
    float y = (v[j] - mean) * inv * g[c] + b[c];
    if (relu) y = y > 0.f ? y : 0.f;
    Y[(long)row * HID + c] = y;
  }
}

// ---------------------------------------------------------------- build kin = [xs_ln[src] || LN(edge_emb)] in bf16 (wave per edge)

__global__ void build_kin_kernel(const float* __restrict__ xs_ln, const float* __restrict__ eemb,
                                 const float* __restrict__ ge, const float* __restrict__ be,
                                 const int* __restrict__ src, __bf16* __restrict__ kin, int E) {
  int wave = threadIdx.x >> 5, lane = threadIdx.x & 31;
  int e = blockIdx.x * 8 + wave;
  if (e >= E) return;
  int sidx = src[e];
  const float* xr = xs_ln + (long)sidx * HID;
  __bf16* kr = kin + (long)e * 256;
#pragma unroll
  for (int j = 0; j < 4; j++) {
    int c = lane + 32 * j;
    kr[c] = (__bf16)xr[c];
  }
  const float* er = eemb + (long)e * HID;
  float v[4];
  float s = 0.f, ss = 0.f;
#pragma unroll
  for (int j = 0; j < 4; j++) {
    v[j] = er[lane + 32 * j];
    s += v[j]; ss += v[j] * v[j];
  }
  s = warp_sum(s); ss = warp_sum(ss);
  float mean = s * (1.0f / 128.0f);
  float inv  = rsqrtf(ss * (1.0f / 128.0f) - mean * mean + 1e-5f);
#pragma unroll
  for (int j = 0; j < 4; j++) {
    int c = lane + 32 * j;
    kr[128 + c] = (__bf16)((v[j] - mean) * inv * ge[c] + be[c]);
  }
}

// ---------------------------------------------------------------- per-edge attention scores + segment max (wave per edge)

__global__ void scores_kernel(const float* __restrict__ q, const float* __restrict__ k,
                              const int* __restrict__ dst, float* __restrict__ sbuf,
                              float* __restrict__ smax, int E) {
  int wave = threadIdx.x >> 5, lane = threadIdx.x & 31;
  int e = blockIdx.x * 8 + wave;
  if (e >= E) return;
  int d = dst[e];
  const float* qr = q + (long)d * HID;
  const float* kr = k + (long)e * HID;
  int c0 = lane * 4;
  v4f qv = *(const v4f*)(qr + c0);
  v4f kv = *(const v4f*)(kr + c0);
  float p = qv[0] * kv[0] + qv[1] * kv[1] + qv[2] * kv[2] + qv[3] * kv[3];
  p += __shfl_xor(p, 1, 32);
  p += __shfl_xor(p, 2, 32);
  if ((lane & 3) == 0) {
    int h = lane >> 2;
    float sv = p * 0.25f;  // 1/sqrt(D=16)
    sbuf[(long)e * 8 + h] = sv;
    atomicMaxFloat(&smax[(long)d * 8 + h], sv);
  }
}

// ---------------------------------------------------------------- exp + segment sum (thread per edge*head)

__global__ void expsum_kernel(const int* __restrict__ dst, float* __restrict__ sbuf,
                              const float* __restrict__ smax, float* __restrict__ den, long EH) {
  long i = (long)blockIdx.x * blockDim.x + threadIdx.x;
  if (i >= EH) return;
  long e = i >> 3; int h = (int)(i & 7);
  int d = dst[e];
  float ex = __expf(sbuf[i] - smax[(long)d * 8 + h]);
  sbuf[i] = ex;
  atomicAdd(&den[(long)d * 8 + h], ex);
}

// ---------------------------------------------------------------- weighted aggregation (wave per edge)

__global__ void agg_kernel(const float* __restrict__ sbuf, const float* __restrict__ den,
                           const float* __restrict__ vbuf, const int* __restrict__ dst,
                           float* __restrict__ agg, int E) {
  int wave = threadIdx.x >> 5, lane = threadIdx.x & 31;
  int e = blockIdx.x * 8 + wave;
  if (e >= E) return;
  int d = dst[e];
#pragma unroll
  for (int j = 0; j < 4; j++) {
    int c = lane + 32 * j;
    int h = c >> 4;
    float alpha = sbuf[(long)e * 8 + h] / (den[(long)d * 8 + h] + 1e-9f);
    atomicAdd(&agg[(long)d * HID + c], alpha * vbuf[(long)e * HID + c]);
  }
}

// ---------------------------------------------------------------- WMMA GEMM (K compile-time, fully unrolled)
// C[M,N] = act(gather(A)[M,KC] @ W[KC,N] + bias + res); W pre-packed bf16 (pack_w_kernel).
// Wave computes a 16x64 tile: 4x v_wmma_f32_16x16x32_bf16 per 32-wide k-step.
// Full unroll turns all address math into immediate load offsets; loads stay
// unpredicated (row/col clamped), only stores are masked on M/N tails.

template <typename AT, int KC>
__global__ void gemm_wmma_kernel(const AT* __restrict__ A, const int* __restrict__ gather,
                                 const __bf16* __restrict__ Wp, const float* __restrict__ bias,
                                 const float* __restrict__ res, float* __restrict__ C,
                                 int M, int N, int relu) {
  const int lane = threadIdx.x & 31;
  const int wave = threadIdx.x >> 5;
  const int hi   = lane >> 4;
  const int p    = lane & 15;
  const int mBase = blockIdx.x * 128 + wave * 16;
  const int nBase = blockIdx.y * 64;
  if (mBase >= M) return;

  int row = mBase + p;
  if (row >= M) row = M - 1;                 // clamp: duplicate compute, masked store
  long arow = gather ? (long)gather[row] : (long)row;
  const AT* Arow = A + (long)arow * KC;

  int colc[4];
#pragma unroll
  for (int t = 0; t < 4; t++) {
    int col = nBase + t * 16 + p;
    colc[t] = col < N ? col : N - 1;         // clamp for loads
  }
  const __bf16* wbase[4];
#pragma unroll
  for (int t = 0; t < 4; t++) wbase[t] = Wp + (long)colc[t] * 32 + hi * 16;

  v8f acc[4];
#pragma unroll
  for (int t = 0; t < 4; t++)
#pragma unroll
    for (int r = 0; r < 8; r++) acc[t][r] = 0.0f;

#pragma unroll
  for (int kb = 0; kb < KC; kb += 32) {
    // ---- A fragment: 16x32 bf16; lane p = row; one contiguous 16-element run at k = hi*16
    v16bf a;
    if constexpr (sizeof(AT) == 4) {
      v4f f0 = *(const v4f*)(Arow + kb + hi * 16);
      v4f f1 = *(const v4f*)(Arow + kb + hi * 16 + 4);
      v4f f2 = *(const v4f*)(Arow + kb + hi * 16 + 8);
      v4f f3 = *(const v4f*)(Arow + kb + hi * 16 + 12);
#pragma unroll
      for (int h = 0; h < 4; h++) {
        a[h]      = (__bf16)f0[h];
        a[4 + h]  = (__bf16)f1[h];
        a[8 + h]  = (__bf16)f2[h];
        a[12 + h] = (__bf16)f3[h];
      }
    } else {
      v8bf b0 = *(const v8bf*)(Arow + kb + hi * 16);
      v8bf b1 = *(const v8bf*)(Arow + kb + hi * 16 + 8);
#pragma unroll
      for (int h = 0; h < 8; h++) {
        a[h]     = b0[h];
        a[8 + h] = b1[h];
      }
    }
    const long kslab = (long)(kb >> 5) * 32 * N;
#pragma unroll
    for (int t = 0; t < 4; t++) {
      // ---- B fragment: 16 contiguous bf16 (32B) from packed weights
      v16bf b = *(const v16bf*)(wbase[t] + kslab);
      acc[t] = __builtin_amdgcn_wmma_f32_16x16x32_bf16(
          false, a, false, b, (short)0, acc[t], false, false);
    }
  }

#pragma unroll
  for (int t = 0; t < 4; t++) {
    int col = nBase + t * 16 + p;
    if (col >= N) continue;
#pragma unroll
    for (int r = 0; r < 8; r++) {
      int orow = mBase + r + hi * 8;
      if (orow >= M) continue;
      float o = acc[t][r];
      if (bias) o += bias[col];
      if (res)  o += res[(long)orow * N + col];
      if (relu) o = o > 0.f ? o : 0.f;
      C[(long)orow * N + col] = o;
    }
  }
}

// ================================================================ host side

struct MLPp {
  const float *W1, *W2, *b1, *b2, *bn, *g;
  const __bf16* W2p;
};
struct GAp {
  const float *Wf1, *Wf2, *Wk, *Wo, *Wq, *Wv, *b2, *b_d, *b_e, *b_s,
              *bf1, *bf2, *bk, *bo, *bq, *bv, *g2, *g_d, *g_e, *g_s;
  const __bf16 *Wf1p, *Wf2p, *Wkp, *Wop, *Wqp, *Wvp;
};

static MLPp parse_mlp(void* const* d, int& i) {
  MLPp p{};
  p.W1 = (const float*)d[i++]; p.W2 = (const float*)d[i++];
  p.b1 = (const float*)d[i++]; p.b2 = (const float*)d[i++];
  p.bn = (const float*)d[i++]; p.g  = (const float*)d[i++];
  return p;
}
static GAp parse_ga(void* const* d, int& i, bool edge) {
  GAp p{};
  p.Wf1 = (const float*)d[i++]; p.Wf2 = (const float*)d[i++];
  p.Wk  = (const float*)d[i++]; p.Wo  = (const float*)d[i++];
  p.Wq  = (const float*)d[i++]; p.Wv  = (const float*)d[i++];
  p.b2  = (const float*)d[i++]; p.b_d = (const float*)d[i++];
  if (edge) p.b_e = (const float*)d[i++];
  p.b_s = (const float*)d[i++];
  p.bf1 = (const float*)d[i++]; p.bf2 = (const float*)d[i++];
  p.bk  = (const float*)d[i++]; p.bo  = (const float*)d[i++];
  p.bq  = (const float*)d[i++]; p.bv  = (const float*)d[i++];
  p.g2  = (const float*)d[i++]; p.g_d = (const float*)d[i++];
  if (edge) p.g_e = (const float*)d[i++];
  p.g_s = (const float*)d[i++];
  return p;
}

struct WS {
  float *m, *t_feat, *h_e, *a_e;
  float *xs_ln, *xd_ln, *q, *kbuf, *vbuf, *sbuf, *smax, *den, *agg, *yln, *ffnh, *etmp;
  __bf16* kin;
};

struct Packer { char* base; size_t off; hipStream_t st; };

static const __bf16* packw(Packer& P, const float* W, int K, int N) {
  __bf16* dst = (__bf16*)(P.base + P.off);
  P.off += ((size_t)K * N * 2 + 255) & ~(size_t)255;
  long n = (long)K * N;
  pack_w_kernel<<<(unsigned)((n + 255) / 256), 256, 0, P.st>>>(W, dst, K, N);
  return dst;
}
static void pack_ga(Packer& P, GAp& p, bool edge) {
  int fin = edge ? 2 * HID : HID;
  p.Wqp  = packw(P, p.Wq,  HID, HID);
  p.Wkp  = packw(P, p.Wk,  fin, HID);
  p.Wvp  = packw(P, p.Wv,  fin, HID);
  p.Wop  = packw(P, p.Wo,  HID, HID);
  p.Wf1p = packw(P, p.Wf1, HID, 4 * HID);
  p.Wf2p = packw(P, p.Wf2, 4 * HID, HID);
}

template <typename AT>
static void gemm_any(const AT* A, const int* gth, const __bf16* Wp, const float* bias,
                     const float* res, float* C, int M, int K, int N, int relu,
                     hipStream_t st) {
  dim3 g((M + 127) / 128, (N + 63) / 64);
  if (K == 128)
    gemm_wmma_kernel<AT, 128><<<g, 256, 0, st>>>(A, gth, Wp, bias, res, C, M, N, relu);
  else if (K == 256)
    gemm_wmma_kernel<AT, 256><<<g, 256, 0, st>>>(A, gth, Wp, bias, res, C, M, N, relu);
  else
    gemm_wmma_kernel<AT, 512><<<g, 256, 0, st>>>(A, gth, Wp, bias, res, C, M, N, relu);
}
static void gemm_f(const float* A, const int* gth, const __bf16* Wp, const float* bias,
                   const float* res, float* C, int M, int K, int N, int relu, hipStream_t st) {
  gemm_any<float>(A, gth, Wp, bias, res, C, M, K, N, relu, st);
}
static void gemm_bf(const __bf16* A, const __bf16* Wp, const float* bias,
                    float* C, int M, int K, int N, hipStream_t st) {
  gemm_any<__bf16>(A, nullptr, Wp, bias, nullptr, C, M, K, N, 0, st);
}
static void fill(float* p, float v, long n, hipStream_t st) {
  fill_kernel<<<(unsigned)((n + 255) / 256), 256, 0, st>>>(p, v, n);
}
static void run_ln(const float* X, const float* g, const float* b, float* Y,
                   int rows, int relu, hipStream_t st) {
  ln_kernel<<<(rows + 7) / 8, 256, 0, st>>>(X, g, b, Y, rows, relu);
}

// TwoLayerMLP with small fan-in: Linear -> LN -> ReLU (fused) -> Linear (WMMA)
static void run_mlp2_small(const MLPp& p, const float* X, int rows, int fi, int fo,
                           float* out, float* hidden_tmp, hipStream_t st) {
  mlp_l1_small_kernel<<<(rows + 7) / 8, 256, 0, st>>>(X, fi, p.W1, p.b1, p.g, p.bn,
                                                      hidden_tmp, rows);
  gemm_f(hidden_tmp, nullptr, p.W2p, p.b2, nullptr, out, rows, HID, fo, 0, st);
}

static void run_attn(const GAp& p, const float* x_src, int n_src, float* x_dst, int n_dst,
                     const int* src, const int* dst, int E, const float* edge_emb,
                     WS& w, hipStream_t st) {
  run_ln(x_src, p.g_s, p.b_s, w.xs_ln, n_src, 0, st);
  run_ln(x_dst, p.g_d, p.b_d, w.xd_ln, n_dst, 0, st);
  gemm_f(w.xd_ln, nullptr, p.Wqp, p.bq, nullptr, w.q, n_dst, HID, HID, 0, st);
  if (edge_emb) {
    build_kin_kernel<<<(E + 7) / 8, 256, 0, st>>>(w.xs_ln, edge_emb, p.g_e, p.b_e,
                                                  src, w.kin, E);
    gemm_bf(w.kin, p.Wkp, p.bk, w.kbuf, E, 256, HID, st);
    gemm_bf(w.kin, p.Wvp, p.bv, w.vbuf, E, 256, HID, st);
  } else {
    gemm_f(w.xs_ln, src, p.Wkp, p.bk, nullptr, w.kbuf, E, HID, HID, 0, st);
    gemm_f(w.xs_ln, src, p.Wvp, p.bv, nullptr, w.vbuf, E, HID, HID, 0, st);
  }
  fill(w.smax, -INFINITY, (long)n_dst * 8, st);
  fill(w.den, 0.0f, (long)n_dst * 8, st);
  fill(w.agg, 0.0f, (long)n_dst * HID, st);
  scores_kernel<<<(E + 7) / 8, 256, 0, st>>>(w.q, w.kbuf, dst, w.sbuf, w.smax, E);
  long EH = (long)E * 8;
  expsum_kernel<<<(unsigned)((EH + 255) / 256), 256, 0, st>>>(dst, w.sbuf, w.smax, w.den, EH);
  agg_kernel<<<(E + 7) / 8, 256, 0, st>>>(w.sbuf, w.den, w.vbuf, dst, w.agg, E);
  gemm_f(w.agg, nullptr, p.Wop, p.bo, x_dst, x_dst, n_dst, HID, HID, 0, st);
  run_ln(x_dst, p.g2, p.b2, w.yln, n_dst, 0, st);
  gemm_f(w.yln, nullptr, p.Wf1p, p.bf1, nullptr, w.ffnh, n_dst, HID, 4 * HID, 1, st);
  gemm_f(w.ffnh, nullptr, p.Wf2p, p.bf2, x_dst, x_dst, n_dst, 4 * HID, HID, 0, st);
}

extern "C" void kernel_launch(void* const* d_in, const int* in_sizes, int n_in,
                              void* d_out, int out_size, void* d_ws, size_t ws_size,
                              hipStream_t stream) {
  (void)in_sizes; (void)n_in; (void)out_size; (void)ws_size;
  // ---- params (JAX pytree order: dict keys sorted, lists in order) ----
  int i = 0;
  MLPp a_emb      = parse_mlp(d_in, i);
  GAp  l2m_attn   = parse_ga(d_in, i, true);
  MLPp l2m_emb    = parse_mlp(d_in, i);
  GAp  m2m_a0     = parse_ga(d_in, i, true);
  GAp  m2m_a1     = parse_ga(d_in, i, true);
  MLPp m2m_a_emb  = parse_mlp(d_in, i);
  GAp  m2m_h0     = parse_ga(d_in, i, true);
  GAp  m2m_h1     = parse_ga(d_in, i, true);
  MLPp m2m_h_emb  = parse_mlp(d_in, i);
  GAp  m2m_s0     = parse_ga(d_in, i, false);
  GAp  m2m_s1     = parse_ga(d_in, i, false);
  const float* mode_tokens = (const float*)d_in[i++];
  GAp  t2m_attn   = parse_ga(d_in, i, true);
  MLPp t2m_emb    = parse_mlp(d_in, i);
  MLPp traj       = parse_mlp(d_in, i);
  // ---- data inputs ----
  const float* a_input  = (const float*)d_in[i++];
  const float* l_embs   = (const float*)d_in[i++];
  const float* t2m_ea   = (const float*)d_in[i++];
  const float* l2m_ea   = (const float*)d_in[i++];
  const float* h_ea     = (const float*)d_in[i++];
  const float* a_ea     = (const float*)d_in[i++];
  const int* t2m_src = (const int*)d_in[i++]; const int* t2m_dst = (const int*)d_in[i++];
  const int* l2m_src = (const int*)d_in[i++]; const int* l2m_dst = (const int*)d_in[i++];
  const int* h_src   = (const int*)d_in[i++]; const int* h_dst   = (const int*)d_in[i++];
  const int* a_src   = (const int*)d_in[i++]; const int* a_dst   = (const int*)d_in[i++];
  const int* s_src   = (const int*)d_in[i++]; const int* s_dst   = (const int*)d_in[i++];
  float* out = (float*)d_out;

  // ---- workspace carve-up ----
  char* base = (char*)d_ws;
  size_t off = 0;
  auto take = [&](size_t bytes) -> char* {
    char* p = base + off;
    off += (bytes + 255) & ~(size_t)255;
    return p;
  };
  WS w;
  w.m      = (float*)take((size_t)NTK * HID * 4);
  w.t_feat = (float*)take((size_t)NT * HID * 4);
  w.h_e    = (float*)take((size_t)E_H * HID * 4);
  w.a_e    = (float*)take((size_t)E_A * HID * 4);
  w.xs_ln  = (float*)take((size_t)NTK * HID * 4);
  w.xd_ln  = (float*)take((size_t)NTK * HID * 4);
  w.q      = (float*)take((size_t)NTK * HID * 4);
  w.kin    = (__bf16*)take((size_t)EMAX * 256 * 2);
  w.kbuf   = (float*)take((size_t)EMAX * HID * 4);
  w.vbuf   = (float*)take((size_t)EMAX * HID * 4);
  w.sbuf   = (float*)take((size_t)EMAX * 8 * 4);
  w.smax   = (float*)take((size_t)NTK * 8 * 4);
  w.den    = (float*)take((size_t)NTK * 8 * 4);
  w.agg    = (float*)take((size_t)NTK * HID * 4);
  w.yln    = (float*)take((size_t)NTK * HID * 4);
  w.ffnh   = (float*)take((size_t)NTK * 4 * HID * 4);
  w.etmp   = (float*)take((size_t)EMAX * HID * 4);

  // ---- pre-pack all GEMM weights to bf16 WMMA-B layout (~3.8 MB) ----
  Packer P{base, off, stream};
  pack_ga(P, t2m_attn, true);
  pack_ga(P, l2m_attn, true);
  pack_ga(P, m2m_h0, true);  pack_ga(P, m2m_h1, true);
  pack_ga(P, m2m_a0, true);  pack_ga(P, m2m_a1, true);
  pack_ga(P, m2m_s0, false); pack_ga(P, m2m_s1, false);
  a_emb.W2p     = packw(P, a_emb.W2, HID, HID);
  t2m_emb.W2p   = packw(P, t2m_emb.W2, HID, HID);
  l2m_emb.W2p   = packw(P, l2m_emb.W2, HID, HID);
  m2m_h_emb.W2p = packw(P, m2m_h_emb.W2, HID, HID);
  m2m_a_emb.W2p = packw(P, m2m_a_emb.W2, HID, HID);
  const __bf16* trajW1p = packw(P, traj.W1, HID, HID);
  const __bf16* trajW2p = packw(P, traj.W2, HID, T_FUT * 2);

  // ---- pipeline ----
  run_mlp2_small(a_emb, a_input, NT, 5, HID, w.t_feat, w.kbuf, stream);
  {
    long total = (long)NTK * HID;
    mode_bcast_kernel<<<(unsigned)((total + 255) / 256), 256, 0, stream>>>(mode_tokens, w.m, total);
  }
  run_mlp2_small(t2m_emb, t2m_ea, E_T2M, 4, HID, w.etmp, w.kbuf, stream);
  run_attn(t2m_attn, w.t_feat, NT, w.m, NTK, t2m_src, t2m_dst, E_T2M, w.etmp, w, stream);
  run_mlp2_small(l2m_emb, l2m_ea, E_L2M, 3, HID, w.etmp, w.kbuf, stream);
  run_attn(l2m_attn, l_embs, N_LANE, w.m, NTK, l2m_src, l2m_dst, E_L2M, w.etmp, w, stream);
  run_mlp2_small(m2m_h_emb, h_ea, E_H, 4, HID, w.h_e, w.kbuf, stream);
  run_mlp2_small(m2m_a_emb, a_ea, E_A, 3, HID, w.a_e, w.kbuf, stream);
  run_attn(m2m_h0, w.m, NTK, w.m, NTK, h_src, h_dst, E_H, w.h_e, w, stream);
  run_attn(m2m_a0, w.m, NTK, w.m, NTK, a_src, a_dst, E_A, w.a_e, w, stream);
  run_attn(m2m_s0, w.m, NTK, w.m, NTK, s_src, s_dst, E_S, nullptr, w, stream);
  run_attn(m2m_h1, w.m, NTK, w.m, NTK, h_src, h_dst, E_H, w.h_e, w, stream);
  run_attn(m2m_a1, w.m, NTK, w.m, NTK, a_src, a_dst, E_A, w.a_e, w, stream);
  run_attn(m2m_s1, w.m, NTK, w.m, NTK, s_src, s_dst, E_S, nullptr, w, stream);
  gemm_f(w.m, nullptr, trajW1p, traj.b1, nullptr, w.kbuf, NTK, HID, HID, 0, stream);
  run_ln(w.kbuf, traj.g, traj.bn, w.vbuf, NTK, 1, stream);
  gemm_f(w.vbuf, nullptr, trajW2p, traj.b2, nullptr, out, NTK, HID, T_FUT * 2, 0, stream);
}
